// FrequencySparseAttention_81252191306591
// MI455X (gfx1250) — matrix operand order repaired and checked by reading, not compiled, not run on hardware
//
#include <hip/hip_runtime.h>
#include <stdint.h>

// ---------------------------------------------------------------------------
// FrequencySparseAttention on MI455X (gfx1250, wave32, WMMA).
//   Pass 1: one-time fp32->bf16 conversion; weights pre-transposed to [N,K].
//   Pass 2: WMMA GEMMs, K-step 64, double-buffered async global->LDS staging
//           (ASYNCcnt), epilogue-peeled steady-state loop.
//   Pass 3: flash attention, 64-key tiles, double-buffered async K/V staging,
//           16 WMMAs per tile between barriers.
//   Pass 4: output projection (bf16 in, fp32 out).
// All matrix math uses v_wmma_f32_16x16x32_bf16 with fp32 accumulation.
// ---------------------------------------------------------------------------

typedef __attribute__((ext_vector_type(16))) __bf16 v16bf;
typedef __attribute__((ext_vector_type(8)))  __bf16 v8bf;
typedef __attribute__((ext_vector_type(8)))  float  v8f;

#define D_MODEL 1024
#define NHEAD   16
#define DH      64
#define BATCH   4
#define SEQ     2048
#define MROWS   (BATCH * SEQ)   /* 8192 */

__device__ __forceinline__ unsigned short f32_to_bf16(float f) {
  unsigned u = __builtin_bit_cast(unsigned, f);
  unsigned r = u + 0x7FFFu + ((u >> 16) & 1u);   // round-to-nearest-even
  return (unsigned short)(r >> 16);
}

__device__ __forceinline__ unsigned pack_bf2(float a, float b) {
  return (unsigned)f32_to_bf16(a) | ((unsigned)f32_to_bf16(b) << 16);
}

// Build a 16-element bf16 fragment from two 16B-aligned groups of 8.
__device__ __forceinline__ v16bf load_a16(const unsigned short* p0,
                                          const unsigned short* p1) {
  v8bf lo = *(const v8bf*)p0;
  v8bf hi = *(const v8bf*)p1;
  v16bf r;
#pragma unroll
  for (int i = 0; i < 8; ++i) { r[i] = lo[i]; r[i + 8] = hi[i]; }
  return r;
}

// Async 16B global -> LDS copy (no data VGPRs; tracked by ASYNCcnt).
// LDS aperture flat addresses carry the LDS offset in the low 32 bits.
__device__ __forceinline__ void async_copy_b128(const void* gptr, void* lptr) {
  unsigned           lds = (unsigned)(uintptr_t)lptr;
  unsigned long long ga  = (unsigned long long)(uintptr_t)gptr;
  asm volatile("global_load_async_to_lds_b128 %0, %1, off"
               :: "v"(lds), "v"(ga) : "memory");
}

__device__ __forceinline__ void wait_asynccnt_0() {
  asm volatile("s_wait_asynccnt 0" ::: "memory");
}
__device__ __forceinline__ void wait_asynccnt_8() {
  asm volatile("s_wait_asynccnt 8" ::: "memory");
}

// ---------------------------------------------------------------------------
// One-time fp32 -> bf16 elementwise conversion (8 elems / thread).
// ---------------------------------------------------------------------------
__global__ __launch_bounds__(256)
void cvt_bf16_kernel(const float* __restrict__ src,
                     unsigned short* __restrict__ dst, int n) {
  int i = (blockIdx.x * 256 + threadIdx.x) * 8;
  if (i >= n) return;                       // n is a multiple of 8
  float4 f0 = ((const float4*)(src + i))[0];
  float4 f1 = ((const float4*)(src + i))[1];
  uint4 p = { pack_bf2(f0.x, f0.y), pack_bf2(f0.z, f0.w),
              pack_bf2(f1.x, f1.y), pack_bf2(f1.z, f1.w) };
  *(uint4*)(dst + i) = p;
}

// ---------------------------------------------------------------------------
// One-time weight convert + transpose: W[K,N] f32  ->  Wt[N,K] bf16.
// ---------------------------------------------------------------------------
__global__ __launch_bounds__(256)
void wtrans_bf16_kernel(const float* __restrict__ W,
                        unsigned short* __restrict__ Wt) {
  __shared__ unsigned short tile[64 * 72];
  const int k0 = blockIdx.y * 64, n0 = blockIdx.x * 64;
  const int t = threadIdx.x;
  {
    int r = t >> 2, c0 = (t & 3) * 16;
    const float* g = W + (size_t)(k0 + r) * D_MODEL + n0 + c0;
    float4 a = ((const float4*)g)[0];
    float4 b = ((const float4*)g)[1];
    float4 c = ((const float4*)g)[2];
    float4 d = ((const float4*)g)[3];
    uint4 p0 = { pack_bf2(a.x, a.y), pack_bf2(a.z, a.w),
                 pack_bf2(b.x, b.y), pack_bf2(b.z, b.w) };
    uint4 p1 = { pack_bf2(c.x, c.y), pack_bf2(c.z, c.w),
                 pack_bf2(d.x, d.y), pack_bf2(d.z, d.w) };
    uint4* dst = (uint4*)&tile[r * 72 + c0];
    dst[0] = p0; dst[1] = p1;
  }
  __syncthreads();
  {
    int n = t >> 2, kc0 = (t & 3) * 16;
    unsigned v[8];
#pragma unroll
    for (int i = 0; i < 8; ++i) {
      unsigned lo = tile[(kc0 + 2 * i) * 72 + n];
      unsigned hh = tile[(kc0 + 2 * i + 1) * 72 + n];
      v[i] = lo | (hh << 16);
    }
    uint4 p0 = { v[0], v[1], v[2], v[3] };
    uint4 p1 = { v[4], v[5], v[6], v[7] };
    uint4* d = (uint4*)&Wt[(size_t)(n0 + n) * D_MODEL + k0 + kc0];
    d[0] = p0; d[1] = p1;
  }
}

// ---------------------------------------------------------------------------
// WMMA GEMM:  out = X[M x 1024](bf16) * Wt[1024 x 1024]^T(bf16,[N,K]) + bias
//   mode 0: out bf16 reshaped [B,H,L,Dh]   (Q, K)
//   mode 1: out bf16 reshaped [B,H,Dh,S]   (V transposed, PV operand order)
//   mode 2: out f32 plain row-major        (final projection)
// 128 threads (4 waves), 64x64 tile, K-step 64.  Double-buffered async
// global->LDS staging; last iteration peeled to keep the loop branch-free.
// ---------------------------------------------------------------------------
__global__ __launch_bounds__(128)
void gemm_wmma_kernel(const unsigned short* __restrict__ X,
                      const unsigned short* __restrict__ Wt,
                      const float* __restrict__ bias, void* __restrict__ out,
                      int mode) {
  __shared__ unsigned short Xl[2][64 * 72];
  __shared__ unsigned short Wl[2][64 * 72];

  const int n0   = blockIdx.x * 64;
  const int m0   = blockIdx.y * 64;
  const int t    = threadIdx.x;
  const int lane = t & 31;
  const int wave = t >> 5;
  const int hi   = lane >> 4;
  const int lm   = lane & 15;

  const int r  = t >> 1;            // staging row (0..63)
  const int c0 = (t & 1) * 32;      // staging col

  const unsigned short* gx0 = X  + (size_t)(m0 + r) * D_MODEL + c0;
  const unsigned short* gw0 = Wt + (size_t)(n0 + r) * D_MODEL + c0;

  v8f acc[4] = {};

  auto issue_tile = [&](int k0, int buf) {
#pragma unroll
    for (int q = 0; q < 4; ++q) {
      async_copy_b128(gx0 + k0 + q * 8, &Xl[buf][r * 72 + c0 + q * 8]);
      async_copy_b128(gw0 + k0 + q * 8, &Wl[buf][r * 72 + c0 + q * 8]);
    }
  };

  auto compute_tile = [&](int buf) {
    const unsigned short* xr = &Xl[buf][(wave * 16 + lm) * 72];
    v16bf a0 = load_a16(xr + 8 * hi,      xr + 16 + 8 * hi);
    v16bf a1 = load_a16(xr + 32 + 8 * hi, xr + 48 + 8 * hi);
#pragma unroll
    for (int nt = 0; nt < 4; ++nt) {
      const unsigned short* wr = &Wl[buf][(nt * 16 + lm) * 72 + 8 * hi];
      v16bf b0 = load_a16(wr, wr + 16);
      v16bf b1 = load_a16(wr + 32, wr + 48);
      acc[nt] = __builtin_amdgcn_wmma_f32_16x16x32_bf16(
          false, a0, false, b0, (short)0, acc[nt], false, false);
      acc[nt] = __builtin_amdgcn_wmma_f32_16x16x32_bf16(
          false, a1, false, b1, (short)0, acc[nt], false, false);
    }
  };

  issue_tile(0, 0);
#pragma unroll 1
  for (int it = 0; it < 15; ++it) {      // steady state: overlap load & math
    issue_tile((it + 1) * 64, (it + 1) & 1);
    wait_asynccnt_8();                   // tile it ready; it+1 in flight
    __syncthreads();
    compute_tile(it & 1);
    __syncthreads();                     // protect buffer before next issue
  }
  wait_asynccnt_0();                     // epilogue tile
  __syncthreads();
  compute_tile(1);

  // store (C/D layout: lane = col n, VGPR j = row j + 8*hi)
#pragma unroll
  for (int nt = 0; nt < 4; ++nt) {
    int n = n0 + nt * 16 + lm;
    float bv = bias[n];
#pragma unroll
    for (int j = 0; j < 8; ++j) {
      int m = m0 + wave * 16 + j + 8 * hi;
      float val = acc[nt][j] + bv;
      if (mode == 2) {
        ((float*)out)[(size_t)m * D_MODEL + n] = val;
      } else {
        int b = m >> 11, l = m & (SEQ - 1);
        int h = n >> 6,  dh = n & (DH - 1);
        if (mode == 0)
          ((unsigned short*)out)[(((size_t)(b * NHEAD + h)) * SEQ + l) * DH + dh] =
              f32_to_bf16(val);
        else
          ((unsigned short*)out)[(((size_t)(b * NHEAD + h)) * DH + dh) * SEQ + l] =
              f32_to_bf16(val);
      }
    }
  }
}

// ---------------------------------------------------------------------------
// Flash-style masked attention.  grid = (L/64, B*H), block = 128 (4 waves).
// 64-key tiles, double-buffered async K/V staging.  Per tile: 8 WMMAs for
// S = Q K^T (4 column blocks x Dh/32 chunks), online softmax + energy mask,
// 8 WMMAs for O += P V.  Output written bf16 [B,L,D] for the final GEMM.
// ---------------------------------------------------------------------------
__global__ __launch_bounds__(128)
void attn_wmma_kernel(const unsigned short* __restrict__ Qb,
                      const unsigned short* __restrict__ Kb,
                      const unsigned short* __restrict__ Vt,
                      const float* __restrict__ E,
                      unsigned short* __restrict__ O) {
  __shared__ unsigned short Kl[2][64 * 72];   // K tile [key][dim]
  __shared__ unsigned short Vl[2][64 * 72];   // V tile transposed [dim][key]
  __shared__ unsigned short Pl[4 * 16 * 72];  // per-wave P scratch

  const int t = threadIdx.x;
  const int lane = t & 31, wave = t >> 5;
  const int hi = lane >> 4, lm = lane & 15;
  const int bh = blockIdx.y;
  const int b  = bh >> 4;
  const int h  = bh & (NHEAD - 1);
  const int q0 = blockIdx.x * 64 + wave * 16;
  const float NEG_INF = -__builtin_inff();

  v16bf qa[2];
  {
    const unsigned short* qrow = Qb + ((size_t)bh * SEQ + q0 + lm) * DH;
#pragma unroll
    for (int c = 0; c < 2; ++c)
      qa[c] = load_a16(qrow + 32 * c + 8 * hi, qrow + 32 * c + 16 + 8 * hi);
  }

  const int sr = t >> 1, sc = (t & 1) * 32;   // staging row/col (K and V^T)

  v8f o_acc[4] = {};
  float mrow[8], lsum[8];
#pragma unroll
  for (int j = 0; j < 8; ++j) { mrow[j] = NEG_INF; lsum[j] = 0.0f; }

  auto issue_tile = [&](int s0, int buf) {
    const unsigned short* gk = Kb + ((size_t)bh * SEQ + s0 + sr) * DH + sc;
    const unsigned short* gv = Vt + ((size_t)bh * DH + sr) * SEQ + s0 + sc;
    unsigned short* lk = &Kl[buf][sr * 72 + sc];
    unsigned short* lv = &Vl[buf][sr * 72 + sc];
#pragma unroll
    for (int q = 0; q < 4; ++q) {
      async_copy_b128(gk + q * 8, lk + q * 8);
      async_copy_b128(gv + q * 8, lv + q * 8);
    }
  };

  auto compute_tile = [&](int s0, int buf) {
    // S = Q K^T : 4 column blocks of 16 keys, mask+scale fused in
    v8f sv[4];
#pragma unroll
    for (int blk = 0; blk < 4; ++blk) {
      v8f s = {};
      int key = 16 * blk + lm;
#pragma unroll
      for (int c = 0; c < 2; ++c) {
        const unsigned short* kr = &Kl[buf][key * 72 + 32 * c + 8 * hi];
        v16bf kb = load_a16(kr, kr + 16);
        s = __builtin_amdgcn_wmma_f32_16x16x32_bf16(
            false, qa[c], false, kb, (short)0, s, false, false);
      }
      bool keep = fabsf(E[b * SEQ + s0 + key]) > 0.1f;
#pragma unroll
      for (int j = 0; j < 8; ++j)
        s[j] = keep ? s[j] * 0.125f : NEG_INF;   // 1/sqrt(Dh) + energy mask
      sv[blk] = s;
    }

    // online softmax (row = j + 8*hi; reductions across 16-lane half)
    float alpha[8];
#pragma unroll
    for (int j = 0; j < 8; ++j) {
      float v = fmaxf(fmaxf(sv[0][j], sv[1][j]), fmaxf(sv[2][j], sv[3][j]));
#pragma unroll
      for (int off = 8; off >= 1; off >>= 1)
        v = fmaxf(v, __shfl_xor(v, off, 32));
      float mo = mrow[j];
      float mn = fmaxf(mo, v);
      float a_, p[4];
      if (mn == NEG_INF) {          // whole row masked so far -> nan_to_num
        a_ = 1.0f; p[0] = p[1] = p[2] = p[3] = 0.0f;
      } else {
        a_ = __expf(mo - mn);
#pragma unroll
        for (int blk = 0; blk < 4; ++blk) p[blk] = __expf(sv[blk][j] - mn);
      }
      float rs = (p[0] + p[1]) + (p[2] + p[3]);
#pragma unroll
      for (int off = 8; off >= 1; off >>= 1)
        rs += __shfl_xor(rs, off, 32);
      mrow[j] = mn;
      lsum[j] = lsum[j] * a_ + rs;
      alpha[j] = a_;
      int prow = j + 8 * hi;
#pragma unroll
      for (int blk = 0; blk < 4; ++blk)
        Pl[wave * 1152 + prow * 72 + 16 * blk + lm] = f32_to_bf16(p[blk]);
    }
    asm volatile("s_wait_dscnt 0" ::: "memory");  // P store -> P reload

    // O += P V  (two K=32 key chunks x 4 dim blocks)
    const unsigned short* pr = &Pl[wave * 1152 + lm * 72];
#pragma unroll
    for (int c = 0; c < 2; ++c) {
      v16bf pa = load_a16(pr + 32 * c + 8 * hi, pr + 32 * c + 16 + 8 * hi);
#pragma unroll
      for (int dc = 0; dc < 4; ++dc) {
        const unsigned short* vr = &Vl[buf][(dc * 16 + lm) * 72 + 32 * c + 8 * hi];
        v16bf vb = load_a16(vr, vr + 16);
        v8f o = o_acc[dc];
        if (c == 0) {
#pragma unroll
          for (int j = 0; j < 8; ++j) o[j] *= alpha[j];
        }
        o_acc[dc] = __builtin_amdgcn_wmma_f32_16x16x32_bf16(
            false, pa, false, vb, (short)0, o, false, false);
      }
    }
  };

  issue_tile(0, 0);
#pragma unroll 1
  for (int it = 0; it < 31; ++it) {      // steady state (SEQ/64 = 32 tiles)
    issue_tile((it + 1) * 64, (it + 1) & 1);
    wait_asynccnt_8();
    __syncthreads();
    compute_tile(it * 64, it & 1);
    __syncthreads();
  }
  wait_asynccnt_0();
  __syncthreads();
  compute_tile(31 * 64, 1);

  // normalize (lsum==0 -> nan_to_num 0) and store bf16 [B,L,D]
  float inv[8];
#pragma unroll
  for (int j = 0; j < 8; ++j) inv[j] = (lsum[j] > 0.0f) ? 1.0f / lsum[j] : 0.0f;
#pragma unroll
  for (int dc = 0; dc < 4; ++dc) {
    int dh = dc * 16 + lm;
#pragma unroll
    for (int j = 0; j < 8; ++j) {
      int row = q0 + j + 8 * hi;
      O[((size_t)b * SEQ + row) * D_MODEL + h * DH + dh] =
          f32_to_bf16(o_acc[dc][j] * inv[j]);
    }
  }
}

// ---------------------------------------------------------------------------
extern "C" void kernel_launch(void* const* d_in, const int* in_sizes, int n_in,
                              void* d_out, int out_size, void* d_ws, size_t ws_size,
                              hipStream_t stream) {
  (void)in_sizes; (void)n_in; (void)out_size; (void)ws_size;

  const float* q_x    = (const float*)d_in[0];
  const float* k_x    = (const float*)d_in[1];
  const float* v_x    = (const float*)d_in[2];
  const float* energy = (const float*)d_in[3];
  const float* Wq = (const float*)d_in[4];  const float* bq = (const float*)d_in[5];
  const float* Wk = (const float*)d_in[6];  const float* bk = (const float*)d_in[7];
  const float* Wv = (const float*)d_in[8];  const float* bv = (const float*)d_in[9];
  const float* Wo = (const float*)d_in[10]; const float* bo = (const float*)d_in[11];

  const size_t MB = 1024 * 1024;
  char* ws = (char*)d_ws;
  unsigned short* Xq  = (unsigned short*)(ws);              // 16 MB (reused as Axb)
  unsigned short* Xk  = (unsigned short*)(ws + 16 * MB);    // 16 MB
  unsigned short* Xv  = (unsigned short*)(ws + 32 * MB);    // 16 MB
  unsigned short* Qb  = (unsigned short*)(ws + 48 * MB);    // 16 MB
  unsigned short* Kb  = (unsigned short*)(ws + 64 * MB);    // 16 MB
  unsigned short* Vt  = (unsigned short*)(ws + 80 * MB);    // 16 MB
  unsigned short* WqT = (unsigned short*)(ws + 96 * MB);    // 2 MB
  unsigned short* WkT = (unsigned short*)(ws + 98 * MB);    // 2 MB
  unsigned short* WvT = (unsigned short*)(ws + 100 * MB);   // 2 MB
  unsigned short* WoT = (unsigned short*)(ws + 102 * MB);   // 2 MB
  unsigned short* Axb = Xq;   // attention output reuses Xq (consumed by then)

  const int NACT = MROWS * D_MODEL;   // 8388608

  // Pass 1: conversions (one-time)
  cvt_bf16_kernel<<<NACT / 2048, 256, 0, stream>>>(q_x, Xq, NACT);
  cvt_bf16_kernel<<<NACT / 2048, 256, 0, stream>>>(k_x, Xk, NACT);
  cvt_bf16_kernel<<<NACT / 2048, 256, 0, stream>>>(v_x, Xv, NACT);
  dim3 gt(D_MODEL / 64, D_MODEL / 64);
  wtrans_bf16_kernel<<<gt, 256, 0, stream>>>(Wq, WqT);
  wtrans_bf16_kernel<<<gt, 256, 0, stream>>>(Wk, WkT);
  wtrans_bf16_kernel<<<gt, 256, 0, stream>>>(Wv, WvT);
  wtrans_bf16_kernel<<<gt, 256, 0, stream>>>(Wo, WoT);

  // Pass 2: projections
  dim3 blk(128);
  dim3 gg(D_MODEL / 64, MROWS / 64);
  gemm_wmma_kernel<<<gg, blk, 0, stream>>>(Xq, WqT, bq, (void*)Qb, 0);
  gemm_wmma_kernel<<<gg, blk, 0, stream>>>(Xk, WkT, bk, (void*)Kb, 0);
  gemm_wmma_kernel<<<gg, blk, 0, stream>>>(Xv, WvT, bv, (void*)Vt, 1);

  // Pass 3: attention
  dim3 ga(SEQ / 64, BATCH * NHEAD);
  attn_wmma_kernel<<<ga, blk, 0, stream>>>(Qb, Kb, Vt, energy, Axb);

  // Pass 4: output projection
  gemm_wmma_kernel<<<gg, blk, 0, stream>>>(Axb, WoT, bo, d_out, 2);
}
